// SpatialAttention_47837345743177
// MI455X (gfx1250) — matrix operand order repaired and checked by reading, not compiled, not run on hardware
//
#include <hip/hip_runtime.h>
#include <hip/hip_bf16.h>

// Attention-augmented conv (Bello et al.) for MI455X / gfx1250, wave32 + WMMA.
// B=8, C=256, H=W=32, Nh=8, DK=DV=64, L=H*W=1024, heads=B*Nh=64.
//
// All GEMMs run as v_wmma_f32_16x16x32_f16 with f32 accumulate. Attention is
// fused flash-style (never materializes the 256MB logits tensor in HBM).

typedef __attribute__((ext_vector_type(16))) _Float16 v16h;
typedef __attribute__((ext_vector_type(8)))  _Float16 v8h;
typedef __attribute__((ext_vector_type(8)))  float    v8f;

#define DEVINL __device__ __forceinline__

// ---------------- WMMA fragment loaders (contiguous k order k = 16*hi + e, same in A and B) -------
// A fragment, 16(M)x32(K) f16. Lane l holds row m = m0 + (l&15); with hi=l>>4
// element e: k = hi*16 + e.  Two 16B loads per lane
// from a row-major [M][K] matrix.
DEVINL v16h load_a_frag(const _Float16* A, int lda, int m0, int k0, int lane) {
  int m  = m0 + (lane & 15);
  int hi = lane >> 4;
  const _Float16* p = A + (size_t)m * lda + k0 + hi * 16;
  v8h lo = *(const v8h*)p;
  v8h hh = *(const v8h*)(p + 8);
  v16h r;
#pragma unroll
  for (int e = 0; e < 8; ++e) { r[e] = lo[e]; r[e + 8] = hh[e]; }
  return r;
}

// B fragment, 32(K)x16(N) f16, sourced from Bt stored [N][K] (K contiguous).
// Lane l holds column n = n0 + (l&15); element e: k = (l>>4)*16 + e.
// One 32B load per lane.
DEVINL v16h load_b_frag(const _Float16* Bt, int ldb, int n0, int k0, int lane) {
  int n  = n0 + (lane & 15);
  int hi = lane >> 4;
  return *(const v16h*)(Bt + (size_t)n * ldb + k0 + hi * 16);
}

DEVINL v8f wmma_f16(v16h a, v16h b, v8f c) {
  return __builtin_amdgcn_wmma_f32_16x16x32_f16(false, a, false, b, (short)0, c,
                                                false, false);
}

// ---------------- conversion / transpose pre-passes --------------------------
__global__ void k_f32_to_f16(const float* __restrict__ s, _Float16* __restrict__ d, int n) {
  int i = blockIdx.x * blockDim.x + threadIdx.x;
  if (i < n) d[i] = (_Float16)s[i];
}

// key_rel_{w,h}: [63][64] f32 -> [64][64] f16 with row 63 zero-padded
// (relative index y2-y+31 is in [0,62], so the pad row is never used).
__global__ void k_cvt_rel(const float* __restrict__ s, _Float16* __restrict__ d) {
  int i = blockIdx.x * blockDim.x + threadIdx.x;
  if (i < 64 * 64) d[i] = (i < 63 * 64) ? (_Float16)s[i] : (_Float16)0.f;
}

// src: [b][Cc][1024] f32 -> dst: [b][1024][Cc] f16  (K-contiguous Bt layout)
__global__ void k_transpose_cvt(const float* __restrict__ src, _Float16* __restrict__ dst, int Cc) {
  int s = blockIdx.x, b = blockIdx.y;
  const float* sb = src + (size_t)b * Cc * 1024;
  _Float16* db = dst + ((size_t)b * 1024 + s) * Cc;
  for (int c = threadIdx.x; c < Cc; c += blockDim.x)
    db[c] = (_Float16)sb[(size_t)c * 1024 + s];
}

// ---------------- QKV projections: out[o,s] = sum_c W[o,c]*x[c,s] + bias -----
// mode 0: Q (scale 1/8, store [head][s][d]); 1: K ([head][s][d]); 2: V ([head][d][s])
__global__ void k_proj(const _Float16* __restrict__ W16, const _Float16* __restrict__ Xt,
                       const float* __restrict__ bias, _Float16* __restrict__ out, int mode) {
  int lane = threadIdx.x & 31, wid = threadIdx.x >> 5;
  int b  = blockIdx.z;
  int m0 = blockIdx.y * 128 + wid * 16;   // o tile
  int n0 = blockIdx.x * 64;               // s tile
  const _Float16* Bt = Xt + (size_t)b * 1024 * 256;
  v8f acc[4] = {};
  for (int k0 = 0; k0 < 256; k0 += 32) {
    v16h a = load_a_frag(W16, 256, m0, k0, lane);
#pragma unroll
    for (int t = 0; t < 4; ++t)
      acc[t] = wmma_f16(a, load_b_frag(Bt, 256, n0 + t * 16, k0, lane), acc[t]);
  }
  int hi = lane >> 4, nl = lane & 15;
#pragma unroll
  for (int t = 0; t < 4; ++t) {
    int n = n0 + t * 16 + nl;
#pragma unroll
    for (int j = 0; j < 8; ++j) {
      int m = m0 + hi * 8 + j;
      float v = acc[t][j] + bias[m];
      if (mode == 0) v *= 0.125f;          // DK^-0.5
      int nh = m >> 6, dd = m & 63;
      size_t head = (size_t)b * 8 + nh;
      if (mode < 2) out[(head * 1024 + n) * 64 + dd] = (_Float16)v;
      else          out[(head * 64 + dd) * 1024 + n] = (_Float16)v;
    }
  }
}

// ---------------- relative-logit tables: RW[i,m] = q_i . rel_k[m] ------------
__global__ void k_rel(const _Float16* __restrict__ Q, const _Float16* __restrict__ Rel,
                      _Float16* __restrict__ out) {
  int lane = threadIdx.x & 31, wid = threadIdx.x >> 5;
  int head = blockIdx.y;
  int m0   = blockIdx.x * 128 + wid * 16;  // query-index tile
  const _Float16* A = Q + (size_t)head * 65536;
  v8f acc[4] = {};
  for (int k0 = 0; k0 < 64; k0 += 32) {
    v16h a = load_a_frag(A, 64, m0, k0, lane);
#pragma unroll
    for (int t = 0; t < 4; ++t)
      acc[t] = wmma_f16(a, load_b_frag(Rel, 64, t * 16, k0, lane), acc[t]);
  }
  int hi = lane >> 4, nl = lane & 15;
#pragma unroll
  for (int t = 0; t < 4; ++t) {
    int n = t * 16 + nl;
#pragma unroll
    for (int j = 0; j < 8; ++j) {
      int m = m0 + hi * 8 + j;
      out[(size_t)head * 65536 + (size_t)m * 64 + n] = (_Float16)acc[t][j];
    }
  }
}

// ---------------- fused flash attention per (head, 16-query tile) ------------
// Computes S^T = K.Q^T so the query index lives in the lane index: softmax
// stats are per-lane + one shfl_xor(16); the exp(S)->B-fragment re-layout for
// the P.V WMMA needs only lane-16 shuffles (no LDS).
__global__ void k_attn(const _Float16* __restrict__ Q, const _Float16* __restrict__ Kf,
                       const _Float16* __restrict__ V, const _Float16* __restrict__ RW,
                       const _Float16* __restrict__ RH, float* __restrict__ O) {
  int lane = threadIdx.x & 31, wid = threadIdx.x >> 5;
  int head = blockIdx.y;
  int i0   = blockIdx.x * 128 + wid * 16;
  const _Float16* Qh = Q  + (size_t)head * 65536;
  const _Float16* Kh = Kf + (size_t)head * 65536;
  const _Float16* Vh = V  + (size_t)head * 65536;   // [d][j], j contiguous
  int hi = lane >> 4, nl = lane & 15;
  int i = i0 + nl, x = i >> 5, y = i & 31;
  const _Float16* rwrow = RW + ((size_t)head * 1024 + i) * 64 + (31 - y);
  const _Float16* rhrow = RH + ((size_t)head * 1024 + i) * 64 + (31 - x);

  v16h bq0 = load_b_frag(Qh, 64, i0, 0, lane);   // loop-invariant Q fragments
  v16h bq1 = load_b_frag(Qh, 64, i0, 32, lane);

  v8f oacc[4] = {};
  float mrun = -1e30f, lrun = 0.f;

  for (int j0 = 0; j0 < 1024; j0 += 32) {
    v8f z = {};
    v8f s0 = wmma_f16(load_a_frag(Kh, 64, j0,       0, lane), bq0, z);
    s0     = wmma_f16(load_a_frag(Kh, 64, j0,      32, lane), bq1, s0);
    v8f s1 = wmma_f16(load_a_frag(Kh, 64, j0 + 16,  0, lane), bq0, z);
    s1     = wmma_f16(load_a_frag(Kh, 64, j0 + 16, 32, lane), bq1, s1);

    float p0[8], p1[8], tmax = -1e30f;
#pragma unroll
    for (int r = 0; r < 8; ++r) {
      int ja = j0 + hi * 8 + r, jb = ja + 16;     // key indices of regs
      float va = s0[r] + (float)rwrow[ja & 31] + (float)rhrow[ja >> 5];
      float vb = s1[r] + (float)rwrow[jb & 31] + (float)rhrow[jb >> 5];
      p0[r] = va; p1[r] = vb;
      tmax = fmaxf(tmax, fmaxf(va, vb));
    }
    tmax = fmaxf(tmax, __shfl_xor(tmax, 16, 32));
    float mnew  = fmaxf(mrun, tmax);
    float scale = __expf(mrun - mnew);
    float tsum = 0.f;
#pragma unroll
    for (int r = 0; r < 8; ++r) {
      p0[r] = __expf(p0[r] - mnew);
      p1[r] = __expf(p1[r] - mnew);
      tsum += p0[r] + p1[r];
    }
    tsum += __shfl_xor(tsum, 16, 32);
    lrun = lrun * scale + tsum;
    mrun = mnew;
#pragma unroll
    for (int t = 0; t < 4; ++t) oacc[t] *= scale;

    // Re-layout exp(S^T) (C layout) into a B fragment P^T[j,i] for P.V:
    // lane keeps column i; the missing 8 rows come from lane^16.
    v16h pf;
#pragma unroll
    for (int r = 0; r < 8; ++r) {
      float t0o = __shfl_xor(p0[r], 16, 32);
      float t1o = __shfl_xor(p1[r], 16, 32);
      pf[r]     = (_Float16)(hi ? t1o   : p0[r]);
      pf[r + 8] = (_Float16)(hi ? p1[r] : t0o);
    }
#pragma unroll
    for (int t = 0; t < 4; ++t)   // O^T[d,i] += V[d, j0:j0+32] * P^T
      oacc[t] = wmma_f16(load_a_frag(Vh, 1024, t * 16, j0, lane), pf, oacc[t]);
  }

  float inv = 1.f / lrun;
#pragma unroll
  for (int t = 0; t < 4; ++t) {
    v8f r = oacc[t] * inv;        // 8 consecutive d per lane -> 32B store
    *(v8f*)(O + ((size_t)head * 1024 + i) * 64 + t * 16 + hi * 8) = r;
  }
}

// ---------------- output projection: out[b,d',s] = Wo . attn_ch + bo ---------
__global__ void k_outproj(const _Float16* __restrict__ Wo16, const _Float16* __restrict__ At,
                          const float* __restrict__ bo, float* __restrict__ out) {
  int lane = threadIdx.x & 31, wid = threadIdx.x >> 5;
  int b = blockIdx.y;
  int t = blockIdx.x * 8 + wid;           // 0..63 wave tiles
  int m0 = (t & 3) * 16;                  // d' tile (M=64)
  int n0 = (t >> 2) * 64;                 // spatial tile (N=1024)
  const _Float16* Bt = At + (size_t)b * 1024 * 512;
  v8f acc[4] = {};
  for (int k0 = 0; k0 < 512; k0 += 32) {
    v16h a = load_a_frag(Wo16, 512, m0, k0, lane);
#pragma unroll
    for (int nt = 0; nt < 4; ++nt)
      acc[nt] = wmma_f16(a, load_b_frag(Bt, 512, n0 + nt * 16, k0, lane), acc[nt]);
  }
  int hi = lane >> 4, nl = lane & 15;
#pragma unroll
  for (int nt = 0; nt < 4; ++nt) {
    int n = n0 + nt * 16 + nl;
#pragma unroll
    for (int j = 0; j < 8; ++j) {
      int m = m0 + hi * 8 + j;
      out[(size_t)b * 65536 + (size_t)m * 1024 + n] = acc[nt][j] + bo[m];
    }
  }
}

// ---------------- host launch -----------------------------------------------
extern "C" void kernel_launch(void* const* d_in, const int* in_sizes, int n_in,
                              void* d_out, int out_size, void* d_ws, size_t ws_size,
                              hipStream_t stream) {
  (void)in_sizes; (void)n_in; (void)out_size; (void)ws_size;
  const float* q_x = (const float*)d_in[0];
  const float* k_x = (const float*)d_in[1];
  const float* v_x = (const float*)d_in[2];
  const float* Wq  = (const float*)d_in[3];
  const float* bq  = (const float*)d_in[4];
  const float* Wk  = (const float*)d_in[5];
  const float* bk  = (const float*)d_in[6];
  const float* Wv  = (const float*)d_in[7];
  const float* bv  = (const float*)d_in[8];
  const float* Wo  = (const float*)d_in[9];
  const float* bo  = (const float*)d_in[10];
  const float* krw = (const float*)d_in[11];
  const float* krh = (const float*)d_in[12];

  char* ws = (char*)d_ws;
  size_t off = 0;
  auto take = [&](size_t bytes) -> char* {
    off = (off + 255) & ~(size_t)255;
    char* p = ws + off;
    off += bytes;
    return p;
  };
  // ~81 MB total workspace
  _Float16* xtq   = (_Float16*)take((size_t)8 * 1024 * 256 * 2);
  _Float16* xtk   = (_Float16*)take((size_t)8 * 1024 * 256 * 2);
  _Float16* xtv   = (_Float16*)take((size_t)8 * 1024 * 256 * 2);
  _Float16* wq16  = (_Float16*)take((size_t)512 * 256 * 2);
  _Float16* wk16  = (_Float16*)take((size_t)512 * 256 * 2);
  _Float16* wv16  = (_Float16*)take((size_t)512 * 256 * 2);
  _Float16* wo16  = (_Float16*)take((size_t)64 * 512 * 2);
  _Float16* relw  = (_Float16*)take((size_t)64 * 64 * 2);
  _Float16* relh  = (_Float16*)take((size_t)64 * 64 * 2);
  _Float16* qf16  = (_Float16*)take((size_t)64 * 1024 * 64 * 2);
  _Float16* kf16  = (_Float16*)take((size_t)64 * 1024 * 64 * 2);
  _Float16* vf16  = (_Float16*)take((size_t)64 * 64 * 1024 * 2);
  _Float16* rw16  = (_Float16*)take((size_t)64 * 1024 * 64 * 2);
  _Float16* rh16  = (_Float16*)take((size_t)64 * 1024 * 64 * 2);
  float*    abuf  = (float*)   take((size_t)8 * 512 * 1024 * 4);
  _Float16* at16  = (_Float16*)take((size_t)8 * 1024 * 512 * 2);

  // weight / rel-key conversions
  k_f32_to_f16<<<(512 * 256 + 255) / 256, 256, 0, stream>>>(Wq, wq16, 512 * 256);
  k_f32_to_f16<<<(512 * 256 + 255) / 256, 256, 0, stream>>>(Wk, wk16, 512 * 256);
  k_f32_to_f16<<<(512 * 256 + 255) / 256, 256, 0, stream>>>(Wv, wv16, 512 * 256);
  k_f32_to_f16<<<(64 * 512 + 255) / 256, 256, 0, stream>>>(Wo, wo16, 64 * 512);
  k_cvt_rel<<<16, 256, 0, stream>>>(krw, relw);
  k_cvt_rel<<<16, 256, 0, stream>>>(krh, relh);
  // activation transposes [b][c][s] -> [b][s][c] f16
  k_transpose_cvt<<<dim3(1024, 8), 256, 0, stream>>>(q_x, xtq, 256);
  k_transpose_cvt<<<dim3(1024, 8), 256, 0, stream>>>(k_x, xtk, 256);
  k_transpose_cvt<<<dim3(1024, 8), 256, 0, stream>>>(v_x, xtv, 256);
  // QKV projections (WMMA)
  dim3 pg(16, 4, 8);
  k_proj<<<pg, 256, 0, stream>>>(wq16, xtq, bq, qf16, 0);
  k_proj<<<pg, 256, 0, stream>>>(wk16, xtk, bk, kf16, 1);
  k_proj<<<pg, 256, 0, stream>>>(wv16, xtv, bv, vf16, 2);
  // relative-logit tables (WMMA)
  dim3 rg(8, 64);
  k_rel<<<rg, 256, 0, stream>>>(qf16, relw, rw16);
  k_rel<<<rg, 256, 0, stream>>>(qf16, relh, rh16);
  // fused flash attention (WMMA + online softmax)
  k_attn<<<dim3(8, 64), 256, 0, stream>>>(qf16, kf16, vf16, rw16, rh16, abuf);
  // channel-major reinterpretation (matches reference reshape) + transpose
  k_transpose_cvt<<<dim3(1024, 8), 256, 0, stream>>>(abuf, at16, 512);
  // output projection (WMMA)
  k_outproj<<<dim3(8, 8), 256, 0, stream>>>(wo16, at16, bo, (float*)d_out);
}
